// MultiHeadAttention_86663850099327
// MI455X (gfx1250) — matrix operand order repaired and checked by reading, not compiled, not run on hardware
//
#include <hip/hip_runtime.h>
#include <hip/hip_bf16.h>

typedef __attribute__((ext_vector_type(16))) _Float16 v16h;
typedef __attribute__((ext_vector_type(8)))  _Float16 v8h;
typedef __attribute__((ext_vector_type(4)))  _Float16 v4h;
typedef __attribute__((ext_vector_type(8)))  float    v8f;
typedef __attribute__((ext_vector_type(4)))  unsigned u32x4;
typedef __attribute__((ext_vector_type(8)))  int      i32x8;
typedef __attribute__((ext_vector_type(4)))  int      i32x4;

#define NB   2
#define SEQ  2048
#define EMBD 1024
#define NH   16
#define HD   64
#define MTOT (NB * SEQ)   // 4096

#if defined(__AMDGCN__) && __has_builtin(__builtin_amdgcn_tensor_load_to_lds) && \
    __has_builtin(__builtin_amdgcn_s_wait_tensorcnt)
#define USE_TDM 1
#else
#define USE_TDM 0
#endif

// ---------------------------------------------------------------------------
// helpers
// ---------------------------------------------------------------------------
static __device__ __forceinline__ v16h ld_pair16(const _Float16* p0, const _Float16* p1) {
  v8h a = *(const v8h*)p0;
  v8h b = *(const v8h*)p1;
  v16h r;
#pragma unroll
  for (int i = 0; i < 8; ++i) { r[i] = a[i]; r[i + 8] = b[i]; }
  return r;
}

static __device__ __forceinline__ v16h ld_contig32(const _Float16* p) {
  return ld_pair16(p, p + 8);
}

static __device__ __forceinline__ v8f wmma_f16(v16h a, v16h b, v8f c) {
  return __builtin_amdgcn_wmma_f32_16x16x32_f16(false, a, false, b, (short)0, c,
                                                false, false);
}
// Same op as the (identical) previous WMMA, sharing the A operand -> reuse-A hint
static __device__ __forceinline__ v8f wmma_f16_ra(v16h a, v16h b, v8f c) {
  return __builtin_amdgcn_wmma_f32_16x16x32_f16(false, a, false, b, (short)0, c,
                                                true, false);
}

#if USE_TDM
#define LDS_ADDR32(p) ((unsigned)(unsigned long long)(const void*)(p))
// 2D TDM descriptor (ISA ch.8): DMA a tile1 x tile0 (rows x elems) block of
// 2-byte elements from global (row stride = stride0 elems) into LDS.
// 6-arg toolchain builtin: (u32x4 g0, i32x8 g1, i32x4, i32x4, i32x8, i32 cpol)
static __device__ __forceinline__ void tdm_load_2d(unsigned lds_byte,
                                                   const void* gptr,
                                                   unsigned dim0, unsigned dim1,
                                                   unsigned tile0, unsigned tile1,
                                                   unsigned stride0) {
  unsigned long long ga = (unsigned long long)gptr;
  u32x4 g0;
  g0[0] = 1u;                                   // count=1, no gather
  g0[1] = lds_byte;                             // lds_addr
  g0[2] = (unsigned)ga;                         // global_addr[31:0]
  g0[3] = (unsigned)((ga >> 32) & 0x1FFFFFFu) | (2u << 30);  // addr[56:32]|type=2
  i32x8 g1;
  g1[0] = (int)(1u << 16);                      // wg_mask=0, data_size=1 (2B)
  g1[1] = (int)((dim0 & 0xFFFFu) << 16);        // barrier_addr=0 | dim0 lo16
  g1[2] = (int)((dim0 >> 16) | ((dim1 & 0xFFFFu) << 16));
  g1[3] = (int)((dim1 >> 16) | (tile0 << 16));
  g1[4] = (int)tile1;                           // tile_dim1 | tile_dim2=0
  g1[5] = (int)stride0;                         // dim0_stride[31:0]
  g1[6] = 0;
  g1[7] = 0;
  i32x4 z4 = {0, 0, 0, 0};
  i32x8 z8 = {0, 0, 0, 0, 0, 0, 0, 0};
  __builtin_amdgcn_tensor_load_to_lds(g0, g1, z4, z4, z8, 0);
}
#endif

// ---------------------------------------------------------------------------
// f32 -> f16 cast
// ---------------------------------------------------------------------------
__global__ void __launch_bounds__(256)
cvt_f32_f16(const float* __restrict__ s, _Float16* __restrict__ d, int n) {
  int i = (blockIdx.x * blockDim.x + threadIdx.x) * 4;
  if (i + 3 < n) {
    float4 f = *(const float4*)(s + i);
    v4h h;
    h[0] = (_Float16)f.x; h[1] = (_Float16)f.y;
    h[2] = (_Float16)f.z; h[3] = (_Float16)f.w;
    *(v4h*)(d + i) = h;
  }
}

// ---------------------------------------------------------------------------
// Shared GEMM core: block = 4 waves, 64(M) x 64(N) tile; W tile (64 x 32 k)
// staged in double-buffered LDS by TDM (shared by all 4 waves); A rows stream
// from global (L2-resident) with a software-pipelined register buffer.
// ---------------------------------------------------------------------------
#define GEMM_CORE(A_, W_, acc_)                                                  \
  const _Float16* arow = (A_) + (size_t)(m0 + col) * EMBD + 16 * halfw;          \
  v8f acc_[4] = {};                                                              \
  GEMM_LOOP(W_, acc_)

#if USE_TDM
#define GEMM_LOOP(W_, acc_)                                                      \
  if (wave == 0)                                                                 \
    tdm_load_2d(LDS_ADDR32(&wbuf[0][0]), (W_) + (size_t)n0 * EMBD,               \
                EMBD, EMBD, 32, 64, EMBD);                                       \
  v16h a = ld_pair16(arow, arow + 8);                                            \
  for (int k = 0; k < EMBD; k += 32) {                                           \
    const int buf = (k >> 5) & 1;                                                \
    if (wave == 0) __builtin_amdgcn_s_wait_tensorcnt(0);                         \
    __syncthreads();                                                             \
    if (wave == 0 && k + 32 < EMBD)                                              \
      tdm_load_2d(LDS_ADDR32(&wbuf[buf ^ 1][0]),                                 \
                  (W_) + (size_t)n0 * EMBD + (k + 32),                           \
                  EMBD, EMBD, 32, 64, EMBD);                                     \
    v16h an = a;                                                                 \
    if (k + 32 < EMBD) an = ld_pair16(arow + k + 32, arow + k + 40);             \
    const _Float16* wb = &wbuf[buf][0] + (size_t)col * 32 + 16 * halfw;          \
    acc_[0] = wmma_f16   (a, ld_contig32(wb),           acc_[0]);                \
    acc_[1] = wmma_f16_ra(a, ld_contig32(wb + 16 * 32), acc_[1]);                \
    acc_[2] = wmma_f16_ra(a, ld_contig32(wb + 32 * 32), acc_[2]);                \
    acc_[3] = wmma_f16_ra(a, ld_contig32(wb + 48 * 32), acc_[3]);                \
    a = an;                                                                      \
    __syncthreads();                                                             \
  }
#else
#define GEMM_LOOP(W_, acc_)                                                      \
  const _Float16* w0 = (W_) + (size_t)(n0 + col) * EMBD + 16 * halfw;            \
  const _Float16* w1 = w0 + 16 * EMBD;                                           \
  const _Float16* w2 = w1 + 16 * EMBD;                                           \
  const _Float16* w3 = w2 + 16 * EMBD;                                           \
  v16h a  = ld_pair16(arow, arow + 8);                                           \
  v16h b0 = ld_contig32(w0), b1 = ld_contig32(w1);                               \
  v16h b2 = ld_contig32(w2), b3 = ld_contig32(w3);                               \
  for (int k = 0; k < EMBD - 32; k += 32) {                                      \
    const int kn = k + 32;                                                       \
    v16h an  = ld_pair16(arow + kn, arow + kn + 8);                              \
    v16h bn0 = ld_contig32(w0 + kn);                                             \
    v16h bn1 = ld_contig32(w1 + kn);                                             \
    v16h bn2 = ld_contig32(w2 + kn);                                             \
    v16h bn3 = ld_contig32(w3 + kn);                                             \
    acc_[0] = wmma_f16   (a, b0, acc_[0]);                                       \
    acc_[1] = wmma_f16_ra(a, b1, acc_[1]);                                       \
    acc_[2] = wmma_f16_ra(a, b2, acc_[2]);                                       \
    acc_[3] = wmma_f16_ra(a, b3, acc_[3]);                                       \
    a = an; b0 = bn0; b1 = bn1; b2 = bn2; b3 = bn3;                              \
  }                                                                              \
  acc_[0] = wmma_f16   (a, b0, acc_[0]);                                         \
  acc_[1] = wmma_f16_ra(a, b1, acc_[1]);                                         \
  acc_[2] = wmma_f16_ra(a, b2, acc_[2]);                                         \
  acc_[3] = wmma_f16_ra(a, b3, acc_[3]);
#endif

// ---------------------------------------------------------------------------
// QKV projection: dst[b,h,s,d] = scale * (A @ W^T)
// ---------------------------------------------------------------------------
__global__ void __launch_bounds__(128)
gemm_qkv(const _Float16* __restrict__ A, const _Float16* __restrict__ W,
         _Float16* __restrict__ dst, float scale) {
#if USE_TDM
  __shared__ __align__(16) _Float16 wbuf[2][64 * 32];
#endif
  const int lane  = threadIdx.x & 31;
  const int wave  = threadIdx.x >> 5;
  const int col   = lane & 15;
  const int halfw = lane >> 4;
  const int m0 = blockIdx.y * 64 + wave * 16;
  const int n0 = blockIdx.x * 64;

  GEMM_CORE(A, W, acc)

#pragma unroll
  for (int t = 0; t < 4; ++t) {
#pragma unroll
    for (int j = 0; j < 8; ++j) {
      int m = m0 + j + 8 * halfw;
      int n = n0 + t * 16 + col;
      int bb = m >> 11, ss = m & (SEQ - 1);
      int hh = n >> 6,  dd = n & 63;
      dst[(((size_t)bb * NH + hh) * SEQ + ss) * HD + dd] =
          (_Float16)(acc[t][j] * scale);
    }
  }
}

// ---------------------------------------------------------------------------
// Output projection: out = (A @ W^T) + bias, f32 output
// ---------------------------------------------------------------------------
__global__ void __launch_bounds__(128)
gemm_proj(const _Float16* __restrict__ A, const _Float16* __restrict__ W,
          const float* __restrict__ bias, float* __restrict__ dst) {
#if USE_TDM
  __shared__ __align__(16) _Float16 wbuf[2][64 * 32];
#endif
  const int lane  = threadIdx.x & 31;
  const int wave  = threadIdx.x >> 5;
  const int col   = lane & 15;
  const int halfw = lane >> 4;
  const int m0 = blockIdx.y * 64 + wave * 16;
  const int n0 = blockIdx.x * 64;

  GEMM_CORE(A, W, acc)

#pragma unroll
  for (int t = 0; t < 4; ++t) {
#pragma unroll
    for (int j = 0; j < 8; ++j) {
      int m = m0 + j + 8 * halfw;
      int n = n0 + t * 16 + col;
      dst[(size_t)m * EMBD + n] = acc[t][j] + bias[n];
    }
  }
}

// ---------------------------------------------------------------------------
// Causal flash attention. Block = 4 waves x 16 queries = 64 queries.
// K 32x64 tiles double-buffered in LDS via TDM (DMA of tile i+1 overlaps
// compute on tile i). V tiles staged TRANSPOSED ([d][key]) by the block with
// register-double-buffered global loads, so PV B-fragments are contiguous
// ds_load_b128 pairs. Per-wave online softmax; P reshaped C->A through LDS.
// ---------------------------------------------------------------------------
__global__ void __launch_bounds__(128)
attn_flash(const _Float16* __restrict__ Q, const _Float16* __restrict__ K,
           const _Float16* __restrict__ V, _Float16* __restrict__ O) {
  __shared__ __align__(16) _Float16 kbuf[2][32 * HD];    // [key][d]
  __shared__ __align__(16) _Float16 vbufT[2][HD * 32];   // [d][key]
  __shared__ __align__(16) _Float16 pbuf[4][16 * 32];

  const int tid   = threadIdx.x;
  const int lane  = tid & 31;
  const int wave  = tid >> 5;
  const int col   = lane & 15;
  const int halfw = lane >> 4;
  const int qtB = blockIdx.x & 31;      // 32 blocks of 64 queries
  const int bh  = blockIdx.x >> 5;      // b*NH + h
  const int q0b = qtB * 64;
  const int q0  = q0b + wave * 16;
  const size_t base = (size_t)bh * SEQ * HD;

  // Q A-fragments (16x64 -> two K=32 fragments), pre-scaled by 1/sqrt(D)
  const _Float16* qrow = Q + base + (size_t)(q0 + col) * HD;
  v16h aq0 = ld_pair16(qrow + 16 * halfw,      qrow + 16 * halfw + 8);
  v16h aq1 = ld_pair16(qrow + 32 + 16 * halfw, qrow + 32 + 16 * halfw + 8);

  v8f oacc[4] = {};
  float mrun[8], lrun[8];
#pragma unroll
  for (int j = 0; j < 8; ++j) { mrun[j] = -__builtin_inff(); lrun[j] = 0.f; }

  const int kend = q0b + 64;

  // V staging: thread owns V[kb + vkey][vdseg..vdseg+15] of the upcoming tile
  const int vkey  = tid >> 2;
  const int vdseg = (tid & 3) * 16;
  const _Float16* vsrc = V + base + (size_t)vkey * HD + vdseg;
  v8h vr0 = *(const v8h*)(vsrc);        // tile 0 prefetch into registers
  v8h vr1 = *(const v8h*)(vsrc + 8);

#if USE_TDM
  if (wave == 0)   // prologue: stage K tile 0
    tdm_load_2d(LDS_ADDR32(&kbuf[0][0]), K + base, HD, SEQ, HD, 32, HD);
#endif
  for (int kb = 0; kb < kend; kb += 32) {
    const int buf = (kb >> 5) & 1;

    // ---- scatter this tile's V registers transposed into LDS ----
    {
      _Float16* vt = &vbufT[buf][0];
#pragma unroll
      for (int i = 0; i < 8; ++i) {
        vt[(vdseg + i) * 32 + vkey]     = vr0[i];
        vt[(vdseg + 8 + i) * 32 + vkey] = vr1[i];
      }
    }
#if USE_TDM
    if (wave == 0) __builtin_amdgcn_s_wait_tensorcnt(0);  // K tile landed
#else
    {
      const int idx = tid * 16;   // cooperative K copy (row-major)
      *(v8h*)(&kbuf[buf][0] + idx)     = *(const v8h*)(K + base + (size_t)kb * HD + idx);
      *(v8h*)(&kbuf[buf][0] + idx + 8) = *(const v8h*)(K + base + (size_t)kb * HD + idx + 8);
    }
#endif
    __syncthreads();   // K tile + transposed V visible to all waves

#if USE_TDM
    if (wave == 0 && kb + 32 < kend)                      // overlap next K DMA
      tdm_load_2d(LDS_ADDR32(&kbuf[buf ^ 1][0]),
                  K + base + (size_t)(kb + 32) * HD, HD, SEQ, HD, 32, HD);
#endif
    if (kb + 32 < kend) {                                 // overlap next V loads
      vr0 = *(const v8h*)(vsrc + (size_t)(kb + 32) * HD);
      vr1 = *(const v8h*)(vsrc + (size_t)(kb + 32) * HD + 8);
    }

    if (kb < q0 + 16) {           // wave-uniform: this wave still has work
      // ---- scores: same-A WMMAs adjacent for reuse-A hints ----
      v8f s0 = {}, s1 = {};
      {
        const _Float16* kr0 = &kbuf[buf][0] + (size_t)col * HD + 16 * halfw;
        const _Float16* kr1 = &kbuf[buf][0] + (size_t)(16 + col) * HD + 16 * halfw;
        v16h b00 = ld_contig32(kr0),      b01 = ld_contig32(kr0 + 32);
        v16h b10 = ld_contig32(kr1),      b11 = ld_contig32(kr1 + 32);
        s0 = wmma_f16   (aq0, b00, s0);
        s1 = wmma_f16_ra(aq0, b10, s1);
        s0 = wmma_f16   (aq1, b01, s0);
        s1 = wmma_f16_ra(aq1, b11, s1);
      }

      // ---- causal mask ----
#pragma unroll
      for (int j = 0; j < 8; ++j) {
        int m = q0 + j + 8 * halfw;
        if (kb + col      > m) s0[j] = -__builtin_inff();
        if (kb + 16 + col > m) s1[j] = -__builtin_inff();
      }

      // ---- online softmax ----
      float mnew[8];
#pragma unroll
      for (int j = 0; j < 8; ++j) {
        float t = fmaxf(s0[j], s1[j]);
#pragma unroll
        for (int d = 1; d < 16; d <<= 1) t = fmaxf(t, __shfl_xor(t, d, 32));
        mnew[j] = fmaxf(mrun[j], t);
      }
#pragma unroll
      for (int j = 0; j < 8; ++j) {
        float p0 = __expf(s0[j] - mnew[j]);
        float p1 = __expf(s1[j] - mnew[j]);
        s0[j] = p0; s1[j] = p1;
        float rs = p0 + p1;
#pragma unroll
        for (int d = 1; d < 16; d <<= 1) rs += __shfl_xor(rs, d, 32);
        float alpha = __expf(mrun[j] - mnew[j]);
        lrun[j] = lrun[j] * alpha + rs;
        mrun[j] = mnew[j];
        oacc[0][j] *= alpha; oacc[1][j] *= alpha;
        oacc[2][j] *= alpha; oacc[3][j] *= alpha;
      }

      // ---- P: C-layout -> per-wave LDS -> A-layout ----
      _Float16* pw = &pbuf[wave][0];
#pragma unroll
      for (int j = 0; j < 8; ++j) {
        int m = j + 8 * halfw;
        pw[m * 32 + col]      = (_Float16)s0[j];
        pw[m * 32 + 16 + col] = (_Float16)s1[j];
      }
      const _Float16* prow = pw + col * 32 + 16 * halfw;  // DS ops in-order/wave
      v16h pa = ld_pair16(prow, prow + 8);

      // ---- O += P (16x32) x V (32x64): B-frags contiguous from vbufT ----
#pragma unroll
      for (int t = 0; t < 4; ++t) {
        const _Float16* vb = &vbufT[buf][0] + (size_t)(t * 16 + col) * 32 + 16 * halfw;
        v16h bv = ld_contig32(vb);
        oacc[t] = (t == 0) ? wmma_f16(pa, bv, oacc[t]) : wmma_f16_ra(pa, bv, oacc[t]);
      }
    }
    __syncthreads();   // all reads of this buffer done before its next refill
  }

  // ---- normalize + store O[b*S+s, h*64+d] (f16 row-major [4096,1024]) ----
  const int b = bh >> 4, h = bh & 15;
#pragma unroll
  for (int t = 0; t < 4; ++t) {
#pragma unroll
    for (int j = 0; j < 8; ++j) {
      int srow = q0 + j + 8 * halfw;
      size_t off = ((size_t)b * SEQ + srow) * EMBD + h * HD + t * 16 + col;
      O[off] = (_Float16)(oacc[t][j] / lrun[j]);
    }
  }
}

// ---------------------------------------------------------------------------
// launch
// ---------------------------------------------------------------------------
extern "C" void kernel_launch(void* const* d_in, const int* in_sizes, int n_in,
                              void* d_out, int out_size, void* d_ws, size_t ws_size,
                              hipStream_t stream) {
  const float* x  = (const float*)d_in[0];
  const float* wq = (const float*)d_in[1];
  const float* wk = (const float*)d_in[2];
  const float* wv = (const float*)d_in[3];
  const float* wp = (const float*)d_in[4];
  const float* bp = (const float*)d_in[5];
  float* out = (float*)d_out;

  char* ws = (char*)d_ws;
  _Float16* x16  = (_Float16*)(ws + ((size_t)0  << 20));  //  8 MB
  _Float16* wq16 = (_Float16*)(ws + ((size_t)8  << 20));  //  2 MB each
  _Float16* wk16 = (_Float16*)(ws + ((size_t)10 << 20));
  _Float16* wv16 = (_Float16*)(ws + ((size_t)12 << 20));
  _Float16* wp16 = (_Float16*)(ws + ((size_t)14 << 20));
  _Float16* Qb   = (_Float16*)(ws + ((size_t)16 << 20));  //  8 MB [B,H,S,D]
  _Float16* Kb   = (_Float16*)(ws + ((size_t)24 << 20));
  _Float16* Vb   = (_Float16*)(ws + ((size_t)32 << 20));
  _Float16* Ob   = (_Float16*)(ws + ((size_t)40 << 20));  //  8 MB [B*S, E]

  const int nx = MTOT * EMBD;
  const int nw = EMBD * EMBD;
  cvt_f32_f16<<<nx / 4 / 256, 256, 0, stream>>>(x,  x16,  nx);
  cvt_f32_f16<<<nw / 4 / 256, 256, 0, stream>>>(wq, wq16, nw);
  cvt_f32_f16<<<nw / 4 / 256, 256, 0, stream>>>(wk, wk16, nw);
  cvt_f32_f16<<<nw / 4 / 256, 256, 0, stream>>>(wv, wv16, nw);
  cvt_f32_f16<<<nw / 4 / 256, 256, 0, stream>>>(wp, wp16, nw);

  dim3 gg(EMBD / 64, MTOT / 64);
  gemm_qkv<<<gg, 128, 0, stream>>>(x16, wq16, Qb, 0.125f);  // 1/sqrt(64)
  gemm_qkv<<<gg, 128, 0, stream>>>(x16, wk16, Kb, 1.0f);
  gemm_qkv<<<gg, 128, 0, stream>>>(x16, wv16, Vb, 1.0f);

  attn_flash<<<NB * NH * (SEQ / 64), 128, 0, stream>>>(Qb, Kb, Vb, Ob);

  gemm_proj<<<gg, 128, 0, stream>>>(Ob, wp16, bp, out);
}
